// RWKV_61460982005861
// MI455X (gfx1250) — matrix operand order repaired and checked by reading, not compiled, not run on hardware
//
#include <hip/hip_runtime.h>
#include <cstdint>
#include <cstddef>

// ---------------- problem constants ----------------
#define C_DIM   768
#define T_DIM   512
#define ROWS    1024        // B_ * T_
#define H_DIM   12
#define KHEAD   64
#define DFF_DIM 2688
#define V_DIM   50304
#define L_DIM   6

// ---------------- CDNA5 WMMA types ----------------
typedef __attribute__((ext_vector_type(16))) __bf16 v16bf;
typedef __attribute__((ext_vector_type(8)))  float  v8f;

__device__ __forceinline__ uint16_t f32_to_bf16(float f) {
  union { float f; uint32_t u; } c; c.f = f;
  uint32_t r = c.u + 0x7FFFu + ((c.u >> 16) & 1u);   // round-to-nearest-even
  return (uint16_t)(r >> 16);
}

// Low 32 bits of a generic pointer to LDS == segment-relative LDS byte address
// (ISA 10.2: LDS aperture keeps the offset in addr[31:0]).
__device__ __forceinline__ uint32_t lds_addr_of(const void* p) {
  return (uint32_t)(uintptr_t)p;
}

// GLOBAL_LOAD_ASYNC_TO_LDS_B128: per-lane DMA memory -> LDS, tracked by ASYNCcnt.
#define ASYNC_COPY_B128(ldsaddr, gptr)                                        \
  asm volatile("global_load_async_to_lds_b128 %0, %1, off"                    \
               :: "v"(ldsaddr), "v"(gptr) : "memory")
#define S_WAIT_ASYNCCNT(n)                                                    \
  asm volatile("s_wait_asynccnt " #n ::: "memory")

// ---------------- GEMM epilogue modes ----------------
enum {
  EPI_NONE = 0,        // out f32 = val
  EPI_SIGMOID = 1,     // out f32 = sigmoid(val)
  EPI_RELUSQ_BF16 = 2, // out bf16 = relu(val)^2
  EPI_ADD_RES = 3,     // out f32 = epi0[idx] + val
  EPI_MUL_ADD = 4      // out f32 = epi1[idx] + val * epi0[idx]
};

// ======================================================================
// Tiled bf16 WMMA GEMM:  Out[M,N] = A[M,K] @ B[N,K]^T   (both bf16)
// Block tile 128x128, K-step 32, 256 threads = 8 waves, wave tile 32x64.
// Double-buffered LDS fed by global_load_async_to_lds_b128 (ASYNCcnt).
// ======================================================================
#define BM 128
#define BN 128
#define BK 32

union FragBF { v16bf v; uint4 q[2]; };

__global__ __launch_bounds__(256) void rwkv_gemm_bf16_wmma(
    const uint16_t* __restrict__ A,     // [M,K] bf16 (raw bits)
    const uint16_t* __restrict__ B,     // [N,K] bf16 (raw bits, row-major = B^T tile)
    void*           __restrict__ Out,
    const float*    __restrict__ epi0,  // gate (MUL_ADD) / residual (ADD_RES)
    const float*    __restrict__ epi1,  // residual (MUL_ADD)
    int M, int N, int K, int mode)
{
  __shared__ __align__(32) uint16_t As[2][BM * BK];   // 2 x 8 KB
  __shared__ __align__(32) uint16_t Bs[2][BN * BK];   // 2 x 8 KB

  const int tid  = threadIdx.x;
  const int bM   = blockIdx.y * BM;
  const int bN   = blockIdx.x * BN;
  const int wid  = tid >> 5;
  const int lane = tid & 31;
  const int wm   = (wid & 3) * 32;   // wave M offset in block tile (4 waves)
  const int wn   = (wid >> 2) * 64;  // wave N offset in block tile (2 waves)
  const int half = lane >> 4;        // K sub-range select per ISA fragment layout
  const int l16  = lane & 15;

  // staging split: 128 rows x 2 chunks of 16 bf16 (2 x b128) per tile, per matrix
  const int srow = tid >> 1;
  const int sseg = (tid & 1) * 16;

  const uint16_t* Ag = A + (size_t)(bM + srow) * K + sseg;
  const uint16_t* Bg = B + (size_t)(bN + srow) * K + sseg;
  const uint32_t  Al0 = lds_addr_of(&As[0][srow * BK + sseg]);
  const uint32_t  Bl0 = lds_addr_of(&Bs[0][srow * BK + sseg]);
  const uint32_t  bufstride = lds_addr_of(&As[1][0]) - lds_addr_of(&As[0][0]);

  v8f acc[2][4] = {};

  // issue tile 0 into buffer 0 (4 async b128 per thread per tile)
  ASYNC_COPY_B128(Al0,      Ag);
  ASYNC_COPY_B128(Al0 + 16, Ag + 8);
  ASYNC_COPY_B128(Bl0,      Bg);
  ASYNC_COPY_B128(Bl0 + 16, Bg + 8);

  const int nk = K / BK;
  for (int t = 0; t < nk; ++t) {
    const int cur = t & 1;
    if (t + 1 < nk) {
      // prefetch tile t+1 into the other buffer while tile t computes
      const uint32_t off = (cur ^ 1) ? bufstride : 0u;
      const uint16_t* ag = Ag + (size_t)(t + 1) * BK;
      const uint16_t* bg = Bg + (size_t)(t + 1) * BK;
      ASYNC_COPY_B128(Al0 + off,      ag);
      ASYNC_COPY_B128(Al0 + off + 16, ag + 8);
      ASYNC_COPY_B128(Bl0 + off,      bg);
      ASYNC_COPY_B128(Bl0 + off + 16, bg + 8);
      S_WAIT_ASYNCCNT(4);   // in-order: tile t's 4 copies have landed
    } else {
      S_WAIT_ASYNCCNT(0);
    }
    __syncthreads();

    // ---- 2x4 WMMA over the wave's 32x64 patch ----
    const uint16_t* asb = As[cur];
    const uint16_t* bsb = Bs[cur];
    #pragma unroll
    for (int tm = 0; tm < 2; ++tm) {
      // A fragment 16x32: lanes 0-15 -> M=l16, K{0..7,16..23};
      //                   lanes 16-31 -> M=l16, K{8..15,24..31}
      FragBF a;
      const uint16_t* ap = asb + (wm + tm * 16 + l16) * BK + half * 8;
      a.q[0] = *(const uint4*)(ap);
      a.q[1] = *(const uint4*)(ap + 16);
      #pragma unroll
      for (int tn = 0; tn < 4; ++tn) {
        // B fragment 32x16 (n-major LDS): lanes 0-15 -> K{0..7,16..23} of col l16,
        //                                 lanes 16-31 -> K{8..15,24..31}
        FragBF b;
        const uint16_t* bp = bsb + (wn + tn * 16 + l16) * BK + half * 8;
        b.q[0] = *(const uint4*)(bp);
        b.q[1] = *(const uint4*)(bp + 16);
        acc[tm][tn] = __builtin_amdgcn_wmma_f32_16x16x32_bf16(
            false, a.v, false, b.v, (short)0, acc[tm][tn], false, false);
      }
    }
    __syncthreads();
  }

  // ---- epilogue: C/D layout -> VGPR i holds M = i (+8 for lane half 1) ----
  #pragma unroll
  for (int tm = 0; tm < 2; ++tm) {
    #pragma unroll
    for (int tn = 0; tn < 4; ++tn) {
      const int gr0 = bM + wm + tm * 16 + half * 8;
      const int gc  = bN + wn + tn * 16 + l16;
      #pragma unroll
      for (int i = 0; i < 8; ++i) {
        const size_t idx = (size_t)(gr0 + i) * N + gc;
        float val = acc[tm][tn][i];
        switch (mode) {
          case EPI_NONE:
            ((float*)Out)[idx] = val; break;
          case EPI_SIGMOID:
            ((float*)Out)[idx] = 1.0f / (1.0f + __expf(-val)); break;
          case EPI_RELUSQ_BF16: {
            float tr = val > 0.0f ? val : 0.0f;
            ((uint16_t*)Out)[idx] = f32_to_bf16(tr * tr); break;
          }
          case EPI_ADD_RES:
            ((float*)Out)[idx] = epi0[idx] + val; break;
          case EPI_MUL_ADD:
            ((float*)Out)[idx] = epi1[idx] + val * epi0[idx]; break;
        }
      }
    }
  }
}

// ======================================================================
// One-shot f32 -> bf16 weight conversion (8 elems / thread).
// ======================================================================
__global__ __launch_bounds__(256) void rwkv_cvt_bf16(
    const float* __restrict__ src, uint16_t* __restrict__ dst, int n8)
{
  const int i = blockIdx.x * 256 + threadIdx.x;
  if (i >= n8) return;
  const float4* s4 = (const float4*)src + (size_t)i * 2;
  float4 f0 = s4[0], f1 = s4[1];
  uint16_t t[8];
  t[0] = f32_to_bf16(f0.x); t[1] = f32_to_bf16(f0.y);
  t[2] = f32_to_bf16(f0.z); t[3] = f32_to_bf16(f0.w);
  t[4] = f32_to_bf16(f1.x); t[5] = f32_to_bf16(f1.y);
  t[6] = f32_to_bf16(f1.z); t[7] = f32_to_bf16(f1.w);
  ((uint4*)dst)[i] = *(const uint4*)t;
}

// ======================================================================
// LayerNorm over C=768, one block per row (256 thr x 3 elems).
// tokens!=null -> gather row from embedding table first.
// ======================================================================
__global__ __launch_bounds__(256) void rwkv_ln(
    const float* __restrict__ X, const int* __restrict__ tokens,
    const float* __restrict__ embed,
    const float* __restrict__ s, const float* __restrict__ b,
    float* __restrict__ outF, uint16_t* __restrict__ outB)
{
  const int row = blockIdx.x;
  const int tid = threadIdx.x;
  const float* src = tokens ? (embed + (size_t)tokens[row] * C_DIM)
                            : (X + (size_t)row * C_DIM);
  float v0 = src[tid], v1 = src[tid + 256], v2 = src[tid + 512];

  __shared__ float red[256];
  __shared__ float stat[2];
  red[tid] = v0 + v1 + v2;
  __syncthreads();
  for (int off = 128; off > 0; off >>= 1) {
    if (tid < off) red[tid] += red[tid + off];
    __syncthreads();
  }
  if (tid == 0) stat[0] = red[0] * (1.0f / C_DIM);
  __syncthreads();
  const float m = stat[0];
  const float d0 = v0 - m, d1 = v1 - m, d2 = v2 - m;
  red[tid] = d0 * d0 + d1 * d1 + d2 * d2;
  __syncthreads();
  for (int off = 128; off > 0; off >>= 1) {
    if (tid < off) red[tid] += red[tid + off];
    __syncthreads();
  }
  if (tid == 0) stat[1] = red[0] * (1.0f / C_DIM);
  __syncthreads();
  const float rstd = rsqrtf(stat[1] + 1e-5f);

  const size_t base = (size_t)row * C_DIM;
  const float dv[3] = {d0, d1, d2};
  #pragma unroll
  for (int e = 0; e < 3; ++e) {
    const int i = tid + e * 256;
    const float val = dv[e] * rstd * s[i] + b[i];
    if (outF) outF[base + i] = val;
    if (outB) outB[base + i] = f32_to_bf16(val);
  }
}

// ======================================================================
// Token-shift lerp -> up to 4 bf16 activation buffers (GEMM A operands).
// ======================================================================
__global__ __launch_bounds__(256) void rwkv_lerp_bf16(
    const float* __restrict__ xln,
    const float* __restrict__ t0, const float* __restrict__ t1,
    const float* __restrict__ t2, const float* __restrict__ t3,
    uint16_t* __restrict__ o0, uint16_t* __restrict__ o1,
    uint16_t* __restrict__ o2, uint16_t* __restrict__ o3)
{
  const int row = blockIdx.x;
  const int tid = threadIdx.x;
  const bool first = (row & (T_DIM - 1)) == 0;   // t==0 within batch: shift = 0
  #pragma unroll
  for (int e = 0; e < 3; ++e) {
    const int i = tid + e * 256;
    const size_t idx = (size_t)row * C_DIM + i;
    const float xv = xln[idx];
    const float xs = first ? 0.0f : xln[idx - C_DIM];
    const float d = xs - xv;
    if (o0) o0[idx] = f32_to_bf16(xv + t0[i] * d);
    if (o1) o1[idx] = f32_to_bf16(xv + t1[i] * d);
    if (o2) o2[idx] = f32_to_bf16(xv + t2[i] * d);
    if (o3) o3[idx] = f32_to_bf16(xv + t3[i] * d);
  }
}

// ======================================================================
// WKV recurrence. One block per (b,h); 256 threads; each thread owns 16
// f32 state cells kv[i][j] (j = tid&63, i = (tid>>6)*16 .. +15).
// ======================================================================
__global__ __launch_bounds__(256) void rwkv_wkv(
    const float* __restrict__ r, const float* __restrict__ k,
    const float* __restrict__ v,
    const float* __restrict__ bonus,   // [H,K] (layer slice)
    const float* __restrict__ decay,   // [H,K] (layer slice)
    float* __restrict__ out)
{
  const int bh = blockIdx.x;
  const int bb = bh / H_DIM;
  const int h  = bh % H_DIM;
  const int tid = threadIdx.x;
  const int j  = tid & 63;
  const int ig = tid >> 6;     // 0..3 -> row group of 16

  float kv[16], uu[16], ww[16];
  #pragma unroll
  for (int ii = 0; ii < 16; ++ii) {
    const int i = ig * 16 + ii;
    kv[ii] = 0.0f;
    uu[ii] = bonus[h * KHEAD + i];
    ww[ii] = __expf(-__expf(decay[h * KHEAD + i]));
  }

  __shared__ float sr[KHEAD], sk[KHEAD], sv[KHEAD];
  __shared__ float redo[4][KHEAD];

  for (int t = 0; t < T_DIM; ++t) {
    const size_t rowoff = ((size_t)(bb * T_DIM + t)) * C_DIM + h * KHEAD;
    if (tid < 64)        sr[tid]       = r[rowoff + tid];
    else if (tid < 128)  sk[tid - 64]  = k[rowoff + tid - 64];
    else if (tid < 192)  sv[tid - 128] = v[rowoff + tid - 128];
    __syncthreads();

    const float vt = sv[j];
    float partial = 0.0f;
    #pragma unroll
    for (int ii = 0; ii < 16; ++ii) {
      const int i = ig * 16 + ii;
      const float outer = sk[i] * vt;
      partial += sr[i] * (kv[ii] + uu[ii] * outer);
      kv[ii] = kv[ii] * ww[ii] + outer;
    }
    redo[ig][j] = partial;
    __syncthreads();
    if (ig == 0)
      out[rowoff + j] = redo[0][j] + redo[1][j] + redo[2][j] + redo[3][j];
    __syncthreads();
  }
}

// ======================================================================
// GroupNorm (H=12 groups of 64, eps=6.4e-4) * silu(g) -> bf16 A operand.
// ======================================================================
__global__ __launch_bounds__(256) void rwkv_gn_silu(
    const float* __restrict__ wkv, const float* __restrict__ g,
    const float* __restrict__ gs, const float* __restrict__ gb,
    uint16_t* __restrict__ outb)
{
  const int row = blockIdx.x;
  const int tid = threadIdx.x;
  __shared__ float gsum[H_DIM], gsq[H_DIM];
  if (tid < H_DIM) { gsum[tid] = 0.0f; gsq[tid] = 0.0f; }
  __syncthreads();

  float xv[3];
  #pragma unroll
  for (int e = 0; e < 3; ++e) {
    const int i = tid + e * 256;
    const float x = wkv[(size_t)row * C_DIM + i];
    xv[e] = x;
    atomicAdd(&gsum[i >> 6], x);        // ds_add_f32
    atomicAdd(&gsq[i >> 6], x * x);
  }
  __syncthreads();

  #pragma unroll
  for (int e = 0; e < 3; ++e) {
    const int i = tid + e * 256;
    const int hh = i >> 6;
    const float m = gsum[hh] * (1.0f / KHEAD);
    const float var = gsq[hh] * (1.0f / KHEAD) - m * m;
    const float nx = (xv[e] - m) * rsqrtf(var + 0.00064f) * gs[i] + gb[i];
    const float gv = g[(size_t)row * C_DIM + i];
    const float sil = gv / (1.0f + __expf(-gv));
    outb[(size_t)row * C_DIM + i] = f32_to_bf16(nx * sil);
  }
}

// ======================================================================
// Host driver: full RWKV forward on one stream.
// ======================================================================
extern "C" void kernel_launch(void* const* d_in, const int* in_sizes, int n_in,
                              void* d_out, int out_size, void* d_ws, size_t ws_size,
                              hipStream_t stream)
{
  (void)in_sizes; (void)n_in; (void)out_size; (void)ws_size;

  const int*   tokens    = (const int*)  d_in[0];
  const float* embed     = (const float*)d_in[1];
  const float* emb_ln_s  = (const float*)d_in[2];
  const float* emb_ln_b  = (const float*)d_in[3];
  const float* tm_ln_s   = (const float*)d_in[4];
  const float* tm_ln_b   = (const float*)d_in[5];
  const float* ts_r      = (const float*)d_in[6];
  const float* ts_k      = (const float*)d_in[7];
  const float* ts_v      = (const float*)d_in[8];
  const float* ts_g      = (const float*)d_in[9];
  const float* W_r       = (const float*)d_in[10];
  const float* W_k       = (const float*)d_in[11];
  const float* W_v       = (const float*)d_in[12];
  const float* W_g       = (const float*)d_in[13];
  const float* W_o       = (const float*)d_in[14];
  const float* bonus     = (const float*)d_in[15];
  const float* decay     = (const float*)d_in[16];
  const float* gn_s      = (const float*)d_in[17];
  const float* gn_b      = (const float*)d_in[18];
  const float* cm_ln_s   = (const float*)d_in[19];
  const float* cm_ln_b   = (const float*)d_in[20];
  const float* cm_ts_in  = (const float*)d_in[21];
  const float* cm_ts_g   = (const float*)d_in[22];
  const float* W_in      = (const float*)d_in[23];
  const float* W_out     = (const float*)d_in[24];
  const float* W_gate    = (const float*)d_in[25];
  const float* head_ln_s = (const float*)d_in[26];
  const float* head_ln_b = (const float*)d_in[27];
  const float* W_unembed = (const float*)d_in[28];

  // ---- workspace bump allocator ----
  char* wp = (char*)d_ws;
  auto alloc = [&](size_t bytes) -> char* {
    char* r = wp;
    wp += (bytes + 255) & ~(size_t)255;
    return r;
  };
  const size_t cc  = (size_t)C_DIM * C_DIM;        // 589824
  const size_t fc  = (size_t)DFF_DIM * C_DIM;      // 2064384
  float*    x     = (float*)   alloc((size_t)ROWS * C_DIM * 4);
  float*    xln   = (float*)   alloc((size_t)ROWS * C_DIM * 4);
  float*    buf_r = (float*)   alloc((size_t)ROWS * C_DIM * 4);
  float*    buf_k = (float*)   alloc((size_t)ROWS * C_DIM * 4);
  float*    buf_v = (float*)   alloc((size_t)ROWS * C_DIM * 4);
  float*    buf_g = (float*)   alloc((size_t)ROWS * C_DIM * 4);
  float*    wkvo  = (float*)   alloc((size_t)ROWS * C_DIM * 4);
  float*    gate  = (float*)   alloc((size_t)ROWS * C_DIM * 4);
  uint16_t* A0    = (uint16_t*)alloc((size_t)ROWS * C_DIM * 2);
  uint16_t* A1    = (uint16_t*)alloc((size_t)ROWS * C_DIM * 2);
  uint16_t* A2    = (uint16_t*)alloc((size_t)ROWS * C_DIM * 2);
  uint16_t* A3    = (uint16_t*)alloc((size_t)ROWS * C_DIM * 2);
  uint16_t* Ao    = (uint16_t*)alloc((size_t)ROWS * C_DIM * 2);
  uint16_t* hid   = (uint16_t*)alloc((size_t)ROWS * DFF_DIM * 2);
  // bf16 weight copies (converted once per call)
  uint16_t* Wr_b  = (uint16_t*)alloc(L_DIM * cc * 2);
  uint16_t* Wk_b  = (uint16_t*)alloc(L_DIM * cc * 2);
  uint16_t* Wv_b  = (uint16_t*)alloc(L_DIM * cc * 2);
  uint16_t* Wg_b  = (uint16_t*)alloc(L_DIM * cc * 2);
  uint16_t* Wo_b  = (uint16_t*)alloc(L_DIM * cc * 2);
  uint16_t* Win_b = (uint16_t*)alloc(L_DIM * fc * 2);
  uint16_t* Wout_b= (uint16_t*)alloc(L_DIM * fc * 2);
  uint16_t* Wgt_b = (uint16_t*)alloc(L_DIM * cc * 2);
  uint16_t* Wun_b = (uint16_t*)alloc((size_t)V_DIM * C_DIM * 2);

  auto cvt = [&](const float* src, uint16_t* dst, size_t n) {
    int n8 = (int)(n / 8);
    rwkv_cvt_bf16<<<(n8 + 255) / 256, 256, 0, stream>>>(src, dst, n8);
  };
  cvt(W_r,       Wr_b,  L_DIM * cc);
  cvt(W_k,       Wk_b,  L_DIM * cc);
  cvt(W_v,       Wv_b,  L_DIM * cc);
  cvt(W_g,       Wg_b,  L_DIM * cc);
  cvt(W_o,       Wo_b,  L_DIM * cc);
  cvt(W_in,      Win_b, L_DIM * fc);
  cvt(W_out,     Wout_b,L_DIM * fc);
  cvt(W_gate,    Wgt_b, L_DIM * cc);
  cvt(W_unembed, Wun_b, (size_t)V_DIM * C_DIM);

  auto gemm = [&](const uint16_t* A, const uint16_t* B, void* Out,
                  const float* e0, const float* e1,
                  int M, int N, int K, int mode) {
    dim3 grid(N / BN, M / BM);
    rwkv_gemm_bf16_wmma<<<grid, 256, 0, stream>>>(A, B, Out, e0, e1, M, N, K, mode);
  };

  // x = LN(embed[tokens])
  rwkv_ln<<<ROWS, 256, 0, stream>>>(nullptr, tokens, embed,
                                    emb_ln_s, emb_ln_b, x, nullptr);

  for (int l = 0; l < L_DIM; ++l) {
    const size_t hk = (size_t)H_DIM * KHEAD;

    // ---- time mixer ----
    rwkv_ln<<<ROWS, 256, 0, stream>>>(x, nullptr, nullptr,
                                      tm_ln_s + l * C_DIM, tm_ln_b + l * C_DIM,
                                      xln, nullptr);
    rwkv_lerp_bf16<<<ROWS, 256, 0, stream>>>(xln,
        ts_r + l * C_DIM, ts_k + l * C_DIM, ts_v + l * C_DIM, ts_g + l * C_DIM,
        A0, A1, A2, A3);
    gemm(A0, Wr_b + l * cc, buf_r, nullptr, nullptr, ROWS, C_DIM, C_DIM, EPI_NONE);
    gemm(A1, Wk_b + l * cc, buf_k, nullptr, nullptr, ROWS, C_DIM, C_DIM, EPI_NONE);
    gemm(A2, Wv_b + l * cc, buf_v, nullptr, nullptr, ROWS, C_DIM, C_DIM, EPI_NONE);
    gemm(A3, Wg_b + l * cc, buf_g, nullptr, nullptr, ROWS, C_DIM, C_DIM, EPI_NONE);
    rwkv_wkv<<<2 * H_DIM, 256, 0, stream>>>(buf_r, buf_k, buf_v,
                                            bonus + l * hk, decay + l * hk, wkvo);
    rwkv_gn_silu<<<ROWS, 256, 0, stream>>>(wkvo, buf_g,
                                           gn_s + l * C_DIM, gn_b + l * C_DIM, Ao);
    // x = x + (gn*silu) @ Wo^T
    gemm(Ao, Wo_b + l * cc, x, x, nullptr, ROWS, C_DIM, C_DIM, EPI_ADD_RES);

    // ---- channel mixer ----
    rwkv_ln<<<ROWS, 256, 0, stream>>>(x, nullptr, nullptr,
                                      cm_ln_s + l * C_DIM, cm_ln_b + l * C_DIM,
                                      xln, nullptr);
    rwkv_lerp_bf16<<<ROWS, 256, 0, stream>>>(xln,
        cm_ts_in + l * C_DIM, cm_ts_g + l * C_DIM, nullptr, nullptr,
        A0, A1, nullptr, nullptr);
    // hidden = relu(x_in @ Win^T)^2   (bf16, feeds next GEMM directly)
    gemm(A0, Win_b + l * fc, hid, nullptr, nullptr,
         ROWS, DFF_DIM, C_DIM, EPI_RELUSQ_BF16);
    // gate = sigmoid(x_g @ Wgate^T)
    gemm(A1, Wgt_b + l * cc, gate, nullptr, nullptr,
         ROWS, C_DIM, C_DIM, EPI_SIGMOID);
    // x = x + (hidden @ Wout^T) * gate
    gemm(hid, Wout_b + l * fc, x, gate, x,
         ROWS, C_DIM, DFF_DIM, EPI_MUL_ADD);
  }

  // ---- head: LN -> bf16, then unembed GEMM into d_out ----
  rwkv_ln<<<ROWS, 256, 0, stream>>>(x, nullptr, nullptr,
                                    head_ln_s, head_ln_b, nullptr, A0);
  gemm(A0, Wun_b, d_out, nullptr, nullptr, ROWS, V_DIM, C_DIM, EPI_NONE);
}
